// SegmentVQVAE_78477642433225
// MI455X (gfx1250) — matrix operand order repaired and checked by reading, not compiled, not run on hardware
//
#include <hip/hip_runtime.h>
#include <hip/hip_bf16.h>
#include <math.h>

typedef __attribute__((ext_vector_type(16))) _Float16 v16h;
typedef __attribute__((ext_vector_type(8)))  _Float16 v8h;
typedef __attribute__((ext_vector_type(8)))  float    v8f;

#define B_  16
#define NV_ 2048
#define NL_ 4096
#define E_  16384
#define M_  (B_*NL_)          // 65536 nodes
#define PI_F 3.14159265358979323846f
#define MT_ 4                 // M-tiles (16 rows each) per wave: 64-row stripe

__device__ __forceinline__ int disc(float t, int num, float lo, float hi) {
    float x = (t - lo) / (hi - lo) * (float)num - 0.5f;
    float r = rintf(x);                       // round-half-even, matches jnp.round
    r = fminf(fmaxf(r, 0.0f), (float)(num - 1));
    return (int)r;
}

// ---------------- weight conversion (f32 -> f16, optional K padding) ----------------
__global__ __launch_bounds__(256) void cvt_pad_kernel(const float* __restrict__ src,
                                                      _Float16* __restrict__ dst,
                                                      int N, int K, int Kp) {
    int i = blockIdx.x * blockDim.x + threadIdx.x;
    if (i >= N * Kp) return;
    int n = i / Kp, k = i - n * Kp;
    dst[i] = (k < K) ? (_Float16)src[(size_t)n * K + k] : (_Float16)0.0f;
}

// ---------------- per-segment endpoints ----------------
__global__ __launch_bounds__(256) void seg_pts_kernel(const float* __restrict__ verts,
                                                      const int* __restrict__ segs,
                                                      float* __restrict__ segpts) {
    int i = blockIdx.x * blockDim.x + threadIdx.x;
    if (i >= M_) return;
    int b  = i / NL_;
    int s0 = segs[(size_t)i * 2 + 0];
    int s1 = segs[(size_t)i * 2 + 1];
    const float* v0 = verts + ((size_t)b * NV_ + s0) * 3;
    const float* v1 = verts + ((size_t)b * NV_ + s1) * 3;
    float* o = segpts + (size_t)i * 6;
    o[0] = v0[0]; o[1] = v0[1]; o[2] = v0[2];
    o[3] = v1[0]; o[4] = v1[1]; o[5] = v1[2];
}

// ---------------- per-edge angle embedding scatter ----------------
__global__ __launch_bounds__(256) void edge_angle_kernel(const float* __restrict__ segpts,
                                                         const int* __restrict__ se,
                                                         const float* __restrict__ angle_tab,
                                                         float* __restrict__ acc,
                                                         float* __restrict__ cnt) {
    int i = blockIdx.x * blockDim.x + threadIdx.x;
    if (i >= B_ * E_) return;
    int b  = i / E_;
    int n0 = b * NL_ + se[(size_t)i * 2 + 0];
    int n1 = b * NL_ + se[(size_t)i * 2 + 1];
    const float* p0 = segpts + (size_t)n0 * 6;
    const float* p1 = segpts + (size_t)n1 * 6;
    float ax = p0[3]-p0[0], ay = p0[4]-p0[1], az = p0[5]-p0[2];
    float bx = p1[3]-p1[0], by = p1[4]-p1[1], bz = p1[5]-p1[2];
    float i0 = 1.0f / fmaxf(sqrtf(ax*ax+ay*ay+az*az), 1e-12f);
    float i1 = 1.0f / fmaxf(sqrtf(bx*bx+by*by+bz*bz), 1e-12f);
    float c  = fabsf((ax*bx + ay*by + az*bz) * i0 * i1);
    c = fminf(c, 1.0f);
    float ang = acosf(c);
    int bin = disc(ang, 128, 0.0f, PI_F * 0.5f);
    const float* te = angle_tab + (size_t)bin * 16;
    #pragma unroll
    for (int k = 0; k < 16; ++k) {
        float v = te[k];
        atomicAdd(&acc[(size_t)n0 * 16 + k], v);
        atomicAdd(&acc[(size_t)n1 * 16 + k], v);
    }
    atomicAdd(&cnt[n0], 1.0f);
    atomicAdd(&cnt[n1], 1.0f);
}

// ---------------- graph in-degree counts (dst column) ----------------
__global__ __launch_bounds__(256) void degree_kernel(const int* __restrict__ se,
                                                     float* __restrict__ gcnt) {
    int i = blockIdx.x * blockDim.x + threadIdx.x;
    if (i >= B_ * E_) return;
    int b = i / E_;
    atomicAdd(&gcnt[b * NL_ + se[(size_t)i * 2 + 1]], 1.0f);
}

// ---------------- build packed 656-dim (padded 672) f16 features ----------------
__global__ __launch_bounds__(128) void pack_kernel(const float* __restrict__ segpts,
                                                   const float* __restrict__ coor_tab,
                                                   const float* __restrict__ len_tab,
                                                   const float* __restrict__ dir_tab,
                                                   const float* __restrict__ midp_tab,
                                                   const float* __restrict__ aacc,
                                                   const float* __restrict__ acnt,
                                                   _Float16* __restrict__ packed) {
    int node = blockIdx.x;
    const float* p = segpts + (size_t)node * 6;
    float crd[6] = {p[0], p[1], p[2], p[3], p[4], p[5]};
    float dx = crd[3]-crd[0], dy = crd[4]-crd[1], dz = crd[5]-crd[2];
    float len = sqrtf(dx*dx + dy*dy + dz*dz);
    float il  = 1.0f / fmaxf(len, 1e-12f);
    float dirs[3] = {dx*il, dy*il, dz*il};
    float mids[3] = {(crd[0]+crd[3])*0.5f, (crd[1]+crd[4])*0.5f, (crd[2]+crd[5])*0.5f};
    float den = fmaxf(acnt[node], 1.0f);
    _Float16* out = packed + (size_t)node * 672;
    for (int f = threadIdx.x; f < 672; f += 128) {
        float val;
        if (f < 384) {
            int c = f >> 6, k = f & 63;
            val = coor_tab[disc(crd[c], 128, -1.0f, 1.0f) * 64 + k];
        } else if (f < 400) {
            val = len_tab[disc(len, 128, 0.0f, 2.0f) * 16 + (f - 384)];
        } else if (f < 592) {
            int g = (f - 400) >> 6, k = (f - 400) & 63;
            val = dir_tab[disc(dirs[g], 128, -1.0f, 1.0f) * 64 + k];
        } else if (f < 608) {
            val = aacc[(size_t)node * 16 + (f - 592)] / den;
        } else if (f < 656) {
            int g = (f - 608) >> 4, k = (f - 608) & 15;
            val = midp_tab[disc(mids[g], 128, -1.0f, 1.0f) * 16 + k];
        } else {
            val = 0.0f;
        }
        out[f] = (_Float16)val;
    }
}

// ---------------- WMMA GEMM: C = A1*W1^T (+ A2*W2^T) + bias, optional ReLU ---------
// A: [M,K] f16 row-major. W: [N,K] f16 row-major (so B[k][n] = W[n][k]).
// One wave computes a 64x16 output stripe (4 M-tiles sharing one B fragment per K
// step) via v_wmma_f32_16x16x32_f16 -> 4 WMMA per B load, 4x less weight traffic.
__global__ __launch_bounds__(256) void gemm_wmma_kernel(
        const _Float16* __restrict__ A1, const _Float16* __restrict__ W1, int K1,
        const _Float16* __restrict__ A2, const _Float16* __restrict__ W2, int K2,
        const float* __restrict__ bias, _Float16* __restrict__ out,
        int M, int N, int relu) {
    int wave = blockIdx.x * (blockDim.x >> 5) + (threadIdx.x >> 5);
    int lane = threadIdx.x & 31;
    int ntiles = N >> 4;
    int mgrps  = M >> 6;                  // 64-row stripes
    int total  = mgrps * ntiles;
    if (wave >= total) return;
    int mg = wave / ntiles, nt = wave - mg * ntiles;
    int half = lane >> 4, r = lane & 15;
    int m0 = (mg << 6) + r;               // A-fragment row (first tile) for this lane
    int n  = (nt << 4) + r;               // B-fragment column for this lane
    v8f c[MT_] = {};

    {   // pair 1
        const _Float16* abase = A1 + (size_t)m0 * K1;
        const _Float16* wrow  = W1 + (size_t)n * K1;
        size_t astride = (size_t)16 * K1; // 16 rows between M-tiles
        for (int k0 = 0; k0 < K1; k0 += 32) {
            union { v16h v; v8h h[2]; } b;
            b.h[0] = *(const v8h*)(wrow + k0 + half * 8);         // K = k0+h*8 .. +7
            b.h[1] = *(const v8h*)(wrow + k0 + 16 + half * 8);    // K = k0+16+h*8 ..
            #pragma unroll
            for (int t = 0; t < MT_; ++t) {
                union { v16h v; v8h h[2]; } a;
                const _Float16* arow = abase + (size_t)t * astride;
                a.h[0] = *(const v8h*)(arow + k0 + half * 8);     // K = k0+h*8 .. +7
                a.h[1] = *(const v8h*)(arow + k0 + 16 + half * 8);// K = k0+16+h*8 ..
                c[t] = __builtin_amdgcn_wmma_f32_16x16x32_f16(
                        false, a.v, false, b.v, (short)0, c[t], false, false);
            }
        }
    }
    if (K2 > 0) {   // fused pair 2 (aggr@Wl^T + x@Wr^T in one accumulator)
        const _Float16* abase = A2 + (size_t)m0 * K2;
        const _Float16* wrow  = W2 + (size_t)n * K2;
        size_t astride = (size_t)16 * K2;
        for (int k0 = 0; k0 < K2; k0 += 32) {
            union { v16h v; v8h h[2]; } b;
            b.h[0] = *(const v8h*)(wrow + k0 + half * 8);
            b.h[1] = *(const v8h*)(wrow + k0 + 16 + half * 8);
            #pragma unroll
            for (int t = 0; t < MT_; ++t) {
                union { v16h v; v8h h[2]; } a;
                const _Float16* arow = abase + (size_t)t * astride;
                a.h[0] = *(const v8h*)(arow + k0 + half * 8);
                a.h[1] = *(const v8h*)(arow + k0 + 16 + half * 8);
                c[t] = __builtin_amdgcn_wmma_f32_16x16x32_f16(
                        false, a.v, false, b.v, (short)0, c[t], false, false);
            }
        }
    }
    float bn = bias ? bias[n] : 0.0f;              // column bias (lane col == n)
    #pragma unroll
    for (int t = 0; t < MT_; ++t) {
        int mbase = (mg << 6) + (t << 4) + (half << 3);  // C layout: M = r + 8*half
        #pragma unroll
        for (int rr = 0; rr < 8; ++rr) {
            float v = c[t][rr] + bn;
            if (relu) v = fmaxf(v, 0.0f);
            out[(size_t)(mbase + rr) * N + n] = (_Float16)v;
        }
    }
}

// ---------------- message scatter-sum: accum[dst] += h[src] ----------------
__global__ __launch_bounds__(256) void scatter_kernel(const _Float16* __restrict__ h,
                                                      const int* __restrict__ se,
                                                      float* __restrict__ accum, int din) {
    int edge = blockIdx.x;                   // B_*E_ edges
    int b = edge / E_;
    int s = b * NL_ + se[(size_t)edge * 2 + 0];
    int d = b * NL_ + se[(size_t)edge * 2 + 1];
    const _Float16* hs = h + (size_t)s * din;
    float* ad = accum + (size_t)d * din;
    for (int f = threadIdx.x; f < din; f += blockDim.x)
        atomicAdd(&ad[f], (float)hs[f]);
}

// ---------------- aggr = accum / max(cnt,1) -> f16 ----------------
__global__ __launch_bounds__(256) void aggr_kernel(const float* __restrict__ accum,
                                                   const float* __restrict__ gcnt,
                                                   _Float16* __restrict__ aggrh, int din) {
    int i = blockIdx.x * blockDim.x + threadIdx.x;
    if (i >= M_ * din) return;
    int node = i / din;
    aggrh[i] = (_Float16)(accum[i] / fmaxf(gcnt[node], 1.0f));
}

// ---------------- row L2 normalization (one wave32 per row) ----------------
__global__ __launch_bounds__(256) void rownorm_kernel(_Float16* __restrict__ x,
                                                      int N, float* __restrict__ fout) {
    int row  = blockIdx.x * (blockDim.x >> 5) + (threadIdx.x >> 5);
    int lane = threadIdx.x & 31;
    if (row >= M_) return;
    _Float16* r = x + (size_t)row * N;
    float ss = 0.0f;
    for (int i = lane; i < N; i += 32) { float v = (float)r[i]; ss += v * v; }
    for (int o = 16; o; o >>= 1) ss += __shfl_xor(ss, o, 32);
    float inv = 1.0f / fmaxf(sqrtf(ss), 1e-12f);
    for (int i = lane; i < N; i += 32) {
        float v = (float)r[i] * inv;
        r[i] = (_Float16)v;
        if (fout) fout[(size_t)row * N + i] = v;
    }
}

// ---------------- SiLU + LayerNorm(64) (one wave32 per row, 2 elems/lane) ----------
__global__ __launch_bounds__(256) void silu_ln_kernel(_Float16* __restrict__ x,
                                                      const float* __restrict__ g,
                                                      const float* __restrict__ bta) {
    int row  = blockIdx.x * (blockDim.x >> 5) + (threadIdx.x >> 5);
    int lane = threadIdx.x & 31;
    if (row >= M_) return;
    _Float16* r = x + (size_t)row * 64;
    float a = (float)r[lane], b = (float)r[lane + 32];
    a = a / (1.0f + expf(-a));
    b = b / (1.0f + expf(-b));
    float s = a + b;
    for (int o = 16; o; o >>= 1) s += __shfl_xor(s, o, 32);
    float mean = s * (1.0f / 64.0f);
    float da = a - mean, db = b - mean;
    float vs = da * da + db * db;
    for (int o = 16; o; o >>= 1) vs += __shfl_xor(vs, o, 32);
    float inv = rsqrtf(vs * (1.0f / 64.0f) + 1e-5f);
    r[lane]      = (_Float16)(da * inv * g[lane]      + bta[lane]);
    r[lane + 32] = (_Float16)(db * inv * g[lane + 32] + bta[lane + 32]);
}

// =====================================================================================
extern "C" void kernel_launch(void* const* d_in, const int* in_sizes, int n_in,
                              void* d_out, int out_size, void* d_ws, size_t ws_size,
                              hipStream_t stream) {
    const float* vertices  = (const float*)d_in[0];
    const int*   segments  = (const int*)d_in[1];
    const int*   seg_edges = (const int*)d_in[2];
    // d_in[3], d_in[4]: masks — all true in this problem, unused.
    const float* coor_tab  = (const float*)d_in[5];
    const float* len_tab   = (const float*)d_in[6];
    const float* dir_tab   = (const float*)d_in[7];
    const float* midp_tab  = (const float*)d_in[8];
    const float* angle_tab = (const float*)d_in[9];
    const float* Wproj     = (const float*)d_in[10];
    const float* bproj     = (const float*)d_in[11];
    const float* ln_g      = (const float*)d_in[12];
    const float* ln_b      = (const float*)d_in[13];
    const int DIN[5]  = {192, 64, 128, 256, 256};
    const int DOUT[5] = {64, 128, 256, 256, 384};
    const float *Wp[5], *bp[5], *Wl[5], *bl[5], *Wr[5];
    for (int i = 0; i < 5; ++i) {
        int base = 14 + i * 5;
        Wp[i] = (const float*)d_in[base + 0];
        bp[i] = (const float*)d_in[base + 1];
        Wl[i] = (const float*)d_in[base + 2];
        bl[i] = (const float*)d_in[base + 3];
        Wr[i] = (const float*)d_in[base + 4];
    }

    // -------- workspace layout (256B aligned regions) --------
    char* w = (char*)d_ws;
    size_t off = 0;
    auto alloc = [&](size_t bytes) -> void* {
        off = (off + 255) & ~(size_t)255;
        void* p = w + off; off += bytes; return p;
    };
    _Float16* WprojH = (_Float16*)alloc((size_t)192 * 672 * 2);
    _Float16 *WpH[5], *WlH[5], *WrH[5];
    for (int i = 0; i < 5; ++i) {
        WpH[i] = (_Float16*)alloc((size_t)DIN[i] * DIN[i] * 2);
        WlH[i] = (_Float16*)alloc((size_t)DOUT[i] * DIN[i] * 2);
        WrH[i] = (_Float16*)alloc((size_t)DOUT[i] * DIN[i] * 2);
    }
    float*    segpts  = (float*)alloc((size_t)M_ * 6 * 4);
    float*    aacc    = (float*)alloc((size_t)M_ * 16 * 4);
    float*    acnt    = (float*)alloc((size_t)M_ * 4);
    float*    gcnt    = (float*)alloc((size_t)M_ * 4);
    _Float16* packed  = (_Float16*)alloc((size_t)M_ * 672 * 2);   // 88 MB
    _Float16* xbuf0   = (_Float16*)alloc((size_t)M_ * 384 * 2);
    _Float16* xbuf1   = (_Float16*)alloc((size_t)M_ * 384 * 2);
    _Float16* hbuf    = (_Float16*)alloc((size_t)M_ * 256 * 2);
    _Float16* aggrh   = (_Float16*)alloc((size_t)M_ * 256 * 2);
    float*    accum   = (float*)packed;   // reuse: packed is dead after proj GEMM

    // -------- weights -> f16 --------
    auto cvt = [&](const float* src, _Float16* dst, int N, int K, int Kp) {
        int tot = N * Kp;
        cvt_pad_kernel<<<(tot + 255) / 256, 256, 0, stream>>>(src, dst, N, K, Kp);
    };
    cvt(Wproj, WprojH, 192, 656, 672);
    for (int i = 0; i < 5; ++i) {
        cvt(Wp[i], WpH[i], DIN[i],  DIN[i], DIN[i]);
        cvt(Wl[i], WlH[i], DOUT[i], DIN[i], DIN[i]);
        cvt(Wr[i], WrH[i], DOUT[i], DIN[i], DIN[i]);
    }

    // -------- zero accumulators --------
    hipMemsetAsync(aacc, 0, (size_t)M_ * 16 * 4, stream);
    hipMemsetAsync(acnt, 0, (size_t)M_ * 4, stream);
    hipMemsetAsync(gcnt, 0, (size_t)M_ * 4, stream);

    // -------- geometry + angle + degree + pack --------
    seg_pts_kernel<<<(M_ + 255) / 256, 256, 0, stream>>>(vertices, segments, segpts);
    edge_angle_kernel<<<(B_ * E_ + 255) / 256, 256, 0, stream>>>(segpts, seg_edges,
                                                                 angle_tab, aacc, acnt);
    degree_kernel<<<(B_ * E_ + 255) / 256, 256, 0, stream>>>(seg_edges, gcnt);
    pack_kernel<<<M_, 128, 0, stream>>>(segpts, coor_tab, len_tab, dir_tab, midp_tab,
                                        aacc, acnt, packed);

    // -------- projection GEMM: x = packed @ Wproj^T + bproj --------
    auto gemm = [&](const _Float16* A1, const _Float16* W1, int K1,
                    const _Float16* A2, const _Float16* W2, int K2,
                    const float* bias, _Float16* out, int N, int relu) {
        int tiles = (M_ / 64) * (N / 16);           // 64-row stripes per wave
        gemm_wmma_kernel<<<(tiles + 7) / 8, 256, 0, stream>>>(A1, W1, K1, A2, W2, K2,
                                                              bias, out, M_, N, relu);
    };
    gemm(packed, WprojH, 672, nullptr, nullptr, 0, bproj, xbuf0, 192, 0);

    // -------- 5 SAGE layers --------
    _Float16* cur = xbuf0;
    _Float16* nxt = xbuf1;
    for (int i = 0; i < 5; ++i) {
        int din = DIN[i], dout = DOUT[i];
        // h = relu(x @ Wp^T + bp)
        gemm(cur, WpH[i], din, nullptr, nullptr, 0, bp[i], hbuf, din, 1);
        // scatter-mean over edges
        hipMemsetAsync(accum, 0, (size_t)M_ * din * 4, stream);
        scatter_kernel<<<B_ * E_, 256, 0, stream>>>(hbuf, seg_edges, accum, din);
        aggr_kernel<<<((size_t)M_ * din + 255) / 256, 256, 0, stream>>>(accum, gcnt,
                                                                        aggrh, din);
        // out = aggr @ Wl^T + bl + x @ Wr^T   (fused dual-K WMMA)
        gemm(aggrh, WlH[i], din, cur, WrH[i], din, bl[i], nxt, dout, 0);
        // row L2 normalization; final layer also emits f32 output
        rownorm_kernel<<<(M_ + 7) / 8, 256, 0, stream>>>(
            nxt, dout, (i == 4) ? (float*)d_out : nullptr);
        if (i == 0)
            silu_ln_kernel<<<(M_ + 7) / 8, 256, 0, stream>>>(nxt, ln_g, ln_b);
        _Float16* t = cur; cur = nxt; nxt = t;
    }
}
